// NeighborEmbedding_7533372638047
// MI455X (gfx1250) — matrix operand order repaired and checked by reading, not compiled, not run on hardware
//
#include <hip/hip_runtime.h>

typedef __attribute__((ext_vector_type(16))) __bf16 v16bf;
typedef __attribute__((ext_vector_type(8)))  __bf16 v8bf;
typedef __attribute__((ext_vector_type(8)))  float  v8f;

#define BN_EPS 1e-5f
#define BALL_R2 0.25f

__device__ __forceinline__ void wait_asynccnt0(){
#if __has_builtin(__builtin_amdgcn_s_wait_asynccnt)
  __builtin_amdgcn_s_wait_asynccnt(0);
#else
  asm volatile("s_wait_asynccnt 0x0" ::: "memory");
#endif
}

// ---------------- utility kernels ----------------
__global__ void k_zero(float* p, int n){
  int i = blockIdx.x*blockDim.x + threadIdx.x;
  if (i < n) p[i] = 0.f;
}

__global__ void k_tobf16(const float* __restrict__ s, __bf16* __restrict__ d, int n){
  int i = blockIdx.x*blockDim.x + threadIdx.x;
  if (i < n) d[i] = (__bf16)s[i];
}

// x [B][3][N] -> xyz [B][N][3]
__global__ void k_xyz_transpose(const float* __restrict__ x, float* __restrict__ xyz,
                                int N, int total){
  int i = blockIdx.x*blockDim.x + threadIdx.x;   // over B*N
  if (i >= total) return;
  int b = i / N, n = i % N;
  const float* xb = x + ((size_t)b*3)*N + n;
  float* o = xyz + (size_t)i*3;
  o[0] = xb[0]; o[1] = xb[(size_t)N]; o[2] = xb[(size_t)2*N];
}

// conv1: out[co][p] = sum_{ci<3} w[co][ci] * x[b][ci][n],  p = b*N+n
__global__ void k_conv1(const float* __restrict__ x, const float* __restrict__ w,
                        float* __restrict__ out, int P, int N){
  int i = blockIdx.x*blockDim.x + threadIdx.x;   // over 64*P
  if (i >= 64*P) return;
  int co = i / P, p = i % P;
  int b = p / N, n = p % N;
  const float* xb = x + ((size_t)b*3)*N + n;
  out[(size_t)co*P + p] =
      w[co*3+0]*xb[0] + w[co*3+1]*xb[(size_t)N] + w[co*3+2]*xb[(size_t)2*N];
}

// per-channel sum / sumsq over src [C][P]; stats[2c]=sum, stats[2c+1]=sumsq (atomic f32)
__global__ void k_stats(const float* __restrict__ src, size_t P, float* __restrict__ stats){
  int c = blockIdx.y;
  const float* row = src + (size_t)c*P;
  float s = 0.f, ss = 0.f;
  for (size_t i = (size_t)blockIdx.x*blockDim.x + threadIdx.x; i < P;
       i += (size_t)gridDim.x*blockDim.x){
    float v = row[i]; s += v; ss += v*v;
  }
  __shared__ float sh[256], sh2[256];
  int t = threadIdx.x;
  sh[t]=s; sh2[t]=ss; __syncthreads();
  for (int o=128;o>0;o>>=1){
    if (t<o){ sh[t]+=sh[t+o]; sh2[t]+=sh2[t+o]; }
    __syncthreads();
  }
  if (t==0){
    __hip_atomic_fetch_add(&stats[2*c+0], sh[0],  __ATOMIC_RELAXED, __HIP_MEMORY_SCOPE_AGENT);
    __hip_atomic_fetch_add(&stats[2*c+1], sh2[0], __ATOMIC_RELAXED, __HIP_MEMORY_SCOPE_AGENT);
  }
}

__device__ __forceinline__ void bn_params(const float* stats, int c, float P,
                                          float& mean, float& inv){
  mean = stats[2*c] / P;
  float var = stats[2*c+1] / P - mean*mean;
  inv = rsqrtf(var + BN_EPS);
}

// bn+relu on src [C][P], write transposed bf16 dst [P][C]
__global__ void k_bnrelu_T_bf16(const float* __restrict__ src, const float* __restrict__ stats,
                                int C, size_t P, __bf16* __restrict__ dst){
  size_t i = (size_t)blockIdx.x*blockDim.x + threadIdx.x;
  if (i >= (size_t)C*P) return;
  int c = (int)(i / P); size_t p = i % P;
  float mean, inv; bn_params(stats, c, (float)P, mean, inv);
  float v = (src[i]-mean)*inv; v = v > 0.f ? v : 0.f;
  dst[p*(size_t)C + c] = (__bf16)v;
}

// bn+relu on src [C][P], write transposed f32 dst [P][C]
__global__ void k_bnrelu_T_f32(const float* __restrict__ src, const float* __restrict__ stats,
                               int C, size_t P, float* __restrict__ dst){
  size_t i = (size_t)blockIdx.x*blockDim.x + threadIdx.x;
  if (i >= (size_t)C*P) return;
  int c = (int)(i / P); size_t p = i % P;
  float mean, inv; bn_params(stats, c, (float)P, mean, inv);
  float v = (src[i]-mean)*inv; v = v > 0.f ? v : 0.f;
  dst[p*(size_t)C + c] = v;
}

// bn+relu on src [C][G*32] then max over the 32 samples of each group.
// out_pt [G][C] (point-major, for next-stage gather), out_bcs [B][C][S] (final layout).
__global__ void k_bnrelu_maxpool(const float* __restrict__ src, const float* __restrict__ stats,
                                 int C, int G, float* __restrict__ out_pt,
                                 float* __restrict__ out_bcs, int S){
  int i = blockIdx.x*blockDim.x + threadIdx.x;
  if (i >= C*G) return;
  int c = i / G, g = i % G;
  size_t P = (size_t)G*32;
  float mean, inv; bn_params(stats, c, (float)P, mean, inv);
  const float* row = src + (size_t)c*P + (size_t)g*32;
  float m = 0.f;  // relu output is >= 0
  for (int k=0;k<32;k++){
    float v = (row[k]-mean)*inv; v = v > 0.f ? v : 0.f;
    m = fmaxf(m, v);
  }
  if (out_pt)  out_pt[(size_t)g*C + c] = m;
  if (out_bcs){ int b = g / S, s = g % S; out_bcs[((size_t)b*C + c)*S + s] = m; }
}

// farthest point sampling; one block (256 thr) per batch. Matches jax scan:
// emit far first, then update dist, then argmax (first-max tie-break).
__global__ void k_fps(const float* __restrict__ xyz, int N, int npoint,
                      int* __restrict__ fidx, float* __restrict__ cxyz){
  __shared__ float dist[2048];
  __shared__ float rv[256];
  __shared__ int   ri[256];
  __shared__ int   s_far;
  int b = blockIdx.x, t = threadIdx.x;
  const float* p = xyz + (size_t)b*N*3;
  for (int n=t;n<N;n+=256) dist[n]=1e10f;
  if (t==0) s_far=0;
  __syncthreads();
  for (int s=0;s<npoint;s++){
    int far = s_far;
    float fx=p[far*3+0], fy=p[far*3+1], fz=p[far*3+2];
    if (t==0){
      fidx[b*npoint+s]=far;
      float* o = cxyz + ((size_t)b*npoint+s)*3;
      o[0]=fx; o[1]=fy; o[2]=fz;
    }
    float bv=-1.f; int bi=0;
    for (int n=t;n<N;n+=256){
      float dx=p[n*3]-fx, dy=p[n*3+1]-fy, dz=p[n*3+2]-fz;
      float d = dx*dx+dy*dy+dz*dz;
      float nd = fminf(dist[n], d); dist[n]=nd;
      if (nd > bv){ bv=nd; bi=n; }
    }
    rv[t]=bv; ri[t]=bi;
    __syncthreads();
    for (int o=128;o>0;o>>=1){
      if (t<o && (rv[t+o]>rv[t] || (rv[t+o]==rv[t] && ri[t+o]<ri[t]))){
        rv[t]=rv[t+o]; ri[t]=ri[t+o];
      }
      __syncthreads();
    }
    if (t==0) s_far = ri[0];
    __syncthreads();
  }
}

// ball query: one wave32 per group; ordered append of first 32 in-radius
// indices via ballot + prefix popcount; pad with first index.
__global__ void k_ballgroup(const float* __restrict__ pts, const float* __restrict__ ctr,
                            int N, int S, int* __restrict__ gi){
  int g = blockIdx.x;                 // b*S + s
  int b = g / S;
  int lane = threadIdx.x;             // 0..31
  const float* P = pts + (size_t)b*N*3;
  float cx=ctr[(size_t)g*3+0], cy=ctr[(size_t)g*3+1], cz=ctr[(size_t)g*3+2];
  __shared__ int buf[32];
  int cnt = 0;
  for (int nb=0; nb<N && cnt<32; nb+=32){
    int n = nb + lane;
    bool in = false;
    if (n < N){
      float dx=P[n*3]-cx, dy=P[n*3+1]-cy, dz=P[n*3+2]-cz;
      in = (dx*dx+dy*dy+dz*dz) <= BALL_R2;
    }
    unsigned mask = (unsigned)__ballot(in);          // wave32: low 32 bits
    int pre = __popc(mask & ((1u<<lane)-1u));
    if (in && (cnt+pre) < 32) buf[cnt+pre] = n;
    cnt += __popc(mask);
  }
  __syncthreads();
  int c = cnt < 32 ? cnt : 32;
  int first = buf[0];
  gi[(size_t)g*32 + lane] = (lane < c) ? buf[lane] : first;
}

// build grouped operand: agg[p][0:D] = f[nbr]-f[center], agg[p][D:2D] = f[center] (bf16)
__global__ void k_gather(const float* __restrict__ fpt, const int* __restrict__ gi,
                         const int* __restrict__ fidx, int Np, int D, int S,
                         __bf16* __restrict__ agg, size_t total){
  size_t i = (size_t)blockIdx.x*blockDim.x + threadIdx.x;
  if (i >= total) return;             // total = G*32*D
  int c = (int)(i % (size_t)D); size_t p = i / (size_t)D;
  size_t g = p >> 5;
  int b = (int)(g / (size_t)S);
  int nb = gi[p];
  int ci = fidx[g];
  float center = fpt[((size_t)b*Np + ci)*D + c];
  float val    = fpt[((size_t)b*Np + nb)*D + c];
  agg[p*(size_t)(2*D) + c]     = (__bf16)(val - center);
  agg[p*(size_t)(2*D) + D + c] = (__bf16)center;
}

// ---------------- WMMA GEMM with async global->LDS staging ----------------
// out[m][p] = sum_k W[m][k] * X[p][k];  W [M][K] bf16, X [P][K] bf16, out [M][P] f32.
// blockDim.x = 2*M (M/16 wave32 per block, one 16-row A tile each); grid.x = P/16.
// The shared 16-point B tile (16 x K bf16, contiguous rows of X) is DMA'd once per
// block into LDS via gfx1250 async global->LDS loads (ASYNCcnt), instead of each of
// the M/16 waves redundantly re-loading it from global memory. A fragments stay on
// the direct global path (distinct rows per wave, L2/L0 resident weights).
// Fragment layouts per CDNA5 ISA §7.12.2 (16-bit A 16x32, 16-bit B 32x16, f32 C/D).
template<int K>
__global__ void k_gemm_wmma(const __bf16* __restrict__ W, const __bf16* __restrict__ X,
                            float* __restrict__ out, size_t P){
  __shared__ __align__(16) __bf16 xtile[16*K];   // 16 points x K channels
  int wave = threadIdx.x >> 5;
  int lane = threadIdx.x & 31;
  int half = lane >> 4;                 // lanes 0-15 vs 16-31
  int l16  = lane & 15;

  // ---- stage B tile: wave 0 issues 512B/iter async copies (32 lanes x 16B) ----
  if (wave == 0){
    const char* src = (const char*)(X + (((size_t)blockIdx.x) << 4)*(size_t)K);
    unsigned lbase = (unsigned)(size_t)(void*)&xtile[0];  // LDS offset (low 32 bits)
    #pragma unroll
    for (int q = 0; q < K/16; ++q){
      int byteoff = (q*32 + lane)*16;
      unsigned long long ga = (unsigned long long)(size_t)(src + byteoff);
      unsigned la = lbase + (unsigned)byteoff;
      asm volatile("global_load_async_to_lds_b128 %0, %1, off"
                   :: "v"(la), "v"(ga) : "memory");
    }
    wait_asynccnt0();
  }
  __syncthreads();

  // ---- WMMA main loop: A from global, B from LDS ----
  int mrow = (wave<<4) + l16;           // A row (M)
  int kbA = half ? 8  : 0;              // A: K offsets {kbA..kbA+7, 16+kbA..16+kbA+7}
  int kbB = half ? 8  : 0;              // B: K offsets {kbB..kbB+7, 16+kbB..16+kbB+7}
  const __bf16* wr = W + (size_t)mrow*K;
  const __bf16* xr = &xtile[(size_t)l16*K];
  v8f c = {0.f,0.f,0.f,0.f,0.f,0.f,0.f,0.f};
  #pragma unroll
  for (int k0=0; k0<K; k0+=32){
    v8bf a0 = *(const v8bf*)(wr + k0 + kbA);
    v8bf a1 = *(const v8bf*)(wr + k0 + 16 + kbA);
    v8bf b0 = *(const v8bf*)(xr + k0 + kbB);      // ds_load_b128
    v8bf b1 = *(const v8bf*)(xr + k0 + 16 + kbB); // ds_load_b128
    v16bf A, Bf;
    #pragma unroll
    for (int e=0;e<8;e++){ A[e]=a0[e]; A[e+8]=a1[e]; Bf[e]=b0[e]; Bf[e+8]=b1[e]; }
    c = __builtin_amdgcn_wmma_f32_16x16x32_bf16(false, A, false, Bf, (short)0, c,
                                                false, false);
  }
  int mbase = (wave<<4) + (half<<3);    // D rows: mbase..mbase+7
  size_t col = (((size_t)blockIdx.x) << 4) + l16;
  float* orow = out + (size_t)mbase*P + col;
  #pragma unroll
  for (int r=0;r<8;r++) orow[(size_t)r*P] = c[r];
}

// ---------------- host ----------------
extern "C" void kernel_launch(void* const* d_in, const int* in_sizes, int n_in,
                              void* d_out, int out_size, void* d_ws, size_t ws_size,
                              hipStream_t stream) {
  (void)in_sizes; (void)n_in; (void)out_size; (void)ws_size;
  const float* x   = (const float*)d_in[0];
  const float* wc1 = (const float*)d_in[1];
  const float* wc2 = (const float*)d_in[2];
  const float* w1a = (const float*)d_in[3];
  const float* w1b = (const float*)d_in[4];
  const float* w2a = (const float*)d_in[5];
  const float* w2b = (const float*)d_in[6];

  const int B=32, N0=2048, C0=64;
  const size_t P0 = (size_t)B*N0;           // 65536
  const int S1=512, M1=128, D1=64;
  const size_t G1=(size_t)B*S1, P1=G1*32;   // 16384, 524288
  const int S2=256, M2=256, D2=128;
  const size_t G2=(size_t)B*S2, P2=G2*32;   // 8192, 262144

  // workspace (needs ~541 MiB; big regions are reused across stages)
  char* ws=(char*)d_ws; size_t off=0;
  auto alloc=[&](size_t bytes)->char*{ char* p=ws+off; off += (bytes+255)&~(size_t)255; return p; };

  float* xyz   = (float*)alloc((size_t)B*N0*3*4);
  float* fpt   = (float*)alloc(P0*(size_t)C0*4);       // [B][N0][64] f32 features
  float* f1pt  = (float*)alloc(G1*(size_t)M1*4);       // [B][512][128] f32 stage-1 out
  float* stats = (float*)alloc(512*4);
  __bf16* wc2b = (__bf16*)alloc((size_t)64*64*2);
  __bf16* w1ab = (__bf16*)alloc((size_t)128*128*2);
  __bf16* w1bb = (__bf16*)alloc((size_t)128*128*2);
  __bf16* w2ab = (__bf16*)alloc((size_t)256*256*2);
  __bf16* w2bb = (__bf16*)alloc((size_t)256*256*2);
  int*   fidx1 = (int*)alloc(G1*4);
  float* cxyz1 = (float*)alloc(G1*3*4);
  int*   gi1   = (int*)alloc(P1*4);
  int*   fidx2 = (int*)alloc(G2*4);
  int*   gi2   = (int*)alloc(P2*4);
  char* bigH = alloc((size_t)268435456);   // f32 GEMM outputs (max 256x262144)
  char* bigA = alloc((size_t)134217728);   // bf16 GEMM B operands (agg)
  char* bigB = alloc((size_t)134217728);   // bf16 GEMM B operands (act)

  float*  c0   = (float*)bigH;                           // conv1 raw [64][P0]
  float*  h0   = (float*)(bigH + ((size_t)1<<27));       // conv2 raw [64][P0]
  __bf16* act0 = (__bf16*)bigA;                          // [P0][64]
  float*  h    = (float*)bigH;                           // stage GEMM outputs
  __bf16* agg1 = (__bf16*)bigA;  __bf16* act1 = (__bf16*)bigB;
  __bf16* agg2 = (__bf16*)bigA;  __bf16* act2 = (__bf16*)bigB;

  float* out_coords = (float*)d_out;                     // [B][256][3]
  float* out_f2     = out_coords + (size_t)B*S2*3;       // [B][256][256]

  auto nb = [](size_t total, int bs){ return (unsigned)((total + (size_t)bs - 1)/(size_t)bs); };

  // ---------------- stage 0: conv1 -> bn_relu -> conv2 -> bn_relu ----------------
  k_xyz_transpose<<<nb((size_t)B*N0,256),256,0,stream>>>(x, xyz, N0, B*N0);
  k_conv1<<<nb(64*P0,256),256,0,stream>>>(x, wc1, c0, (int)P0, N0);
  k_zero<<<2,256,0,stream>>>(stats, 512);
  k_stats<<<dim3(64,64),256,0,stream>>>(c0, P0, stats);
  k_bnrelu_T_bf16<<<nb((size_t)64*P0,256),256,0,stream>>>(c0, stats, 64, P0, act0);
  k_tobf16<<<nb(64*64,256),256,0,stream>>>(wc2, wc2b, 64*64);
  k_gemm_wmma<64><<<(unsigned)(P0/16), 128, 0, stream>>>(wc2b, act0, h0, P0);
  k_zero<<<2,256,0,stream>>>(stats, 512);
  k_stats<<<dim3(64,64),256,0,stream>>>(h0, P0, stats);
  k_bnrelu_T_f32<<<nb((size_t)64*P0,256),256,0,stream>>>(h0, stats, 64, P0, fpt);

  // ---------------- stage 1: FPS(512) -> ball(32) -> 128x128 convs -> max ----------------
  k_fps<<<B,256,0,stream>>>(xyz, N0, S1, fidx1, cxyz1);
  k_ballgroup<<<(unsigned)G1,32,0,stream>>>(xyz, cxyz1, N0, S1, gi1);
  k_gather<<<nb(P1*(size_t)D1,256),256,0,stream>>>(fpt, gi1, fidx1, N0, D1, S1, agg1, P1*(size_t)D1);
  k_tobf16<<<nb(128*128,256),256,0,stream>>>(w1a, w1ab, 128*128);
  k_tobf16<<<nb(128*128,256),256,0,stream>>>(w1b, w1bb, 128*128);
  k_gemm_wmma<128><<<(unsigned)(P1/16), 256, 0, stream>>>(w1ab, agg1, h, P1);
  k_zero<<<2,256,0,stream>>>(stats, 512);
  k_stats<<<dim3(64,128),256,0,stream>>>(h, P1, stats);
  k_bnrelu_T_bf16<<<nb((size_t)128*P1,256),256,0,stream>>>(h, stats, 128, P1, act1);
  k_gemm_wmma<128><<<(unsigned)(P1/16), 256, 0, stream>>>(w1bb, act1, h, P1);
  k_zero<<<2,256,0,stream>>>(stats, 512);
  k_stats<<<dim3(64,128),256,0,stream>>>(h, P1, stats);
  k_bnrelu_maxpool<<<nb(G1*(size_t)M1,256),256,0,stream>>>(h, stats, M1, (int)G1, f1pt, nullptr, S1);

  // ---------------- stage 2: FPS(256) -> ball(32) -> 256x256 convs -> max ----------------
  k_fps<<<B,256,0,stream>>>(cxyz1, S1, S2, fidx2, out_coords);
  k_ballgroup<<<(unsigned)G2,32,0,stream>>>(cxyz1, out_coords, S1, S2, gi2);
  k_gather<<<nb(P2*(size_t)D2,256),256,0,stream>>>(f1pt, gi2, fidx2, S1, D2, S2, agg2, P2*(size_t)D2);
  k_tobf16<<<nb(256*256,256),256,0,stream>>>(w2a, w2ab, 256*256);
  k_tobf16<<<nb(256*256,256),256,0,stream>>>(w2b, w2bb, 256*256);
  k_gemm_wmma<256><<<(unsigned)(P2/16), 512, 0, stream>>>(w2ab, agg2, h, P2);
  k_zero<<<2,256,0,stream>>>(stats, 512);
  k_stats<<<dim3(64,256),256,0,stream>>>(h, P2, stats);
  k_bnrelu_T_bf16<<<nb((size_t)256*P2,256),256,0,stream>>>(h, stats, 256, P2, act2);
  k_gemm_wmma<256><<<(unsigned)(P2/16), 512, 0, stream>>>(w2bb, act2, h, P2);
  k_zero<<<2,256,0,stream>>>(stats, 512);
  k_stats<<<dim3(64,256),256,0,stream>>>(h, P2, stats);
  k_bnrelu_maxpool<<<nb(G2*(size_t)M2,256),256,0,stream>>>(h, stats, M2, (int)G2, nullptr, out_f2, S2);
}
